// RSSM_39788577030454
// MI455X (gfx1250) — matrix operand order repaired and checked by reading, not compiled, not run on hardware
//
#include <hip/hip_runtime.h>
#include <hip/hip_bf16.h>
#include <stdint.h>

#define T_STEPS 64
#define B_SIZE  64
#define STOCH   1024
#define DETER   1024
#define NODE    1024
#define EMBED   1024
#define ACT_DIM 64

typedef __bf16 bf16_t;
typedef __bf16 v16bf __attribute__((ext_vector_type(16)));
typedef __bf16 v8bf  __attribute__((ext_vector_type(8)));
typedef float  v8f   __attribute__((ext_vector_type(8)));

// LDS panel: 64 columns x 32 K-values (bf16) per chunk, row stride padded to
// 80B (16B-aligned for async b128 writes; 20-bank stride avoids conflicts).
#define PANEL_ROW   80
#define PANEL_BYTES (64 * PANEL_ROW)   // 5120 per buffer

// ---------------------------------------------------------------------------
// Prologue kernels: weight transpose+convert, plain convert, zero fill
// ---------------------------------------------------------------------------
__global__ void rssm_convert_t(const float* __restrict__ src, bf16_t* __restrict__ dst,
                               int K, int N) {
  int i = blockIdx.x * blockDim.x + threadIdx.x;
  if (i >= K * N) return;
  int n = i / K;
  int k = i - n * K;
  dst[i] = (bf16_t)src[(size_t)k * N + n];   // dst[N,K] = src[K,N]^T
}

__global__ void rssm_convert(const float* __restrict__ src, bf16_t* __restrict__ dst,
                             int n) {
  int i = blockIdx.x * blockDim.x + threadIdx.x;
  if (i < n) dst[i] = (bf16_t)src[i];
}

__global__ void rssm_zero_f32(float* __restrict__ p, int n) {
  int i = blockIdx.x * blockDim.x + threadIdx.x;
  if (i < n) p[i] = 0.0f;
}

__global__ void rssm_zero_bf16(bf16_t* __restrict__ p, int n) {
  int i = blockIdx.x * blockDim.x + threadIdx.x;
  if (i < n) p[i] = (bf16_t)0.0f;
}

// ---------------------------------------------------------------------------
// Async Global->LDS stage of one 64-col x 32-K bf16 weight panel.
// 256 16-byte granules; 128 threads issue 2 async b128 transfers each wave
// (ASYNCcnt += 2 per wave). Row c of the panel = Wt[(bcol0+c)*K + kc .. +31].
// ---------------------------------------------------------------------------
__device__ __forceinline__ void rssm_stage_panel(const bf16_t* Wt, int K, int bcol0,
                                                 int kc, uint32_t lds_base, int tid) {
#pragma unroll
  for (int q = 0; q < 2; ++q) {
    int g      = q * 128 + tid;        // granule 0..255
    int c      = g >> 2;               // panel row (column of C)
    int within = (g & 3) * 16;         // byte offset inside 64B row payload
    unsigned long long ga =
        (unsigned long long)(const unsigned char*)(Wt + (size_t)(bcol0 + c) * K + kc) +
        (unsigned)within;
    uint32_t la = lds_base + (uint32_t)(c * PANEL_ROW + within);
    asm volatile("global_load_async_to_lds_b128 %0, %1, off"
                 :: "v"(la), "v"(ga)
                 : "memory");
  }
}

// ---------------------------------------------------------------------------
// Dual-job WMMA GEMM with double-buffered async-LDS B panels.
//   C[64,N] = act( [A1 | A2][64, K1+K2](bf16) @ Wt[N, K1+K2]^T(bf16) + bias )
// blockDim = 128 (4 waves). wave -> M-tile (M=64 -> 4). blockIdx.x -> 64-col
// N-group; all 4 waves share the LDS-staged B panel (4x less L2 B-traffic),
// each wave runs 4 N-subtiles so one A fragment feeds 4 WMMAs.
// blockIdx.y selects job0/job1 (uniform; EXEC all-ones around WMMA/DS ops).
// K1, K2 must be multiples of 32 (K-chunks never straddle the concat seam).
// ---------------------------------------------------------------------------
struct GemmJob {
  const bf16_t* A1;
  const bf16_t* A2;        // may be null (K2 == 0)
  const bf16_t* Wt;        // [N, K1+K2] row-major
  const float*  bias;      // [N]
  float*        Cf;        // optional f32 output [64, N]
  bf16_t*       Cbf;       // optional bf16 output [64, N]
  int K1, K2, N, elu;
};

__global__ void rssm_gemm2(GemmJob job0, GemmJob job1) {
  const GemmJob j = (blockIdx.y == 0) ? job0 : job1;

  __shared__ unsigned char panel[2][PANEL_BYTES];

  const int tid  = threadIdx.x;
  const int lane = tid & 31;
  const int wave = tid >> 5;
  const int l15  = lane & 15;
  const int hi   = lane >> 4;          // lane half
  const int kb8  = hi * 8;             // A-layout per-half K base
  const int K    = j.K1 + j.K2;
  const int nch  = K / 32;
  const int bcol0 = blockIdx.x * 64;

  const int arow = wave * 16 + l15;
  const bf16_t* A1p = j.A1 + (size_t)arow * j.K1;
  const bf16_t* A2p = j.A2 ? (j.A2 + (size_t)arow * j.K2) : (const bf16_t*)nullptr;

  // raw LDS byte offsets of the two panel buffers (low 32 bits of generic addr)
  const uint32_t lds0 = (uint32_t)(uintptr_t)&panel[0][0];
  const uint32_t lds1 = (uint32_t)(uintptr_t)&panel[1][0];

  int bcol[4];
  v8f  c[4];
#pragma unroll
  for (int n = 0; n < 4; ++n) {
    bcol[n] = bcol0 + n * 16 + l15;
    const float bv = j.bias[bcol[n]];
#pragma unroll
    for (int d = 0; d < 8; ++d) c[n][d] = bv;
  }

  // prime the pipeline: stage chunk 0 into buffer 0
  rssm_stage_panel(j.Wt, K, bcol0, 0, lds0, tid);

  for (int i = 0; i < nch; ++i) {
    const int kc = i * 32;

    // A fragment loads issued early (latency hidden behind stage/barrier):
    // halves 0..7 = K kc+kb8..+7 ; halves 8..15 = K kc+16+kb8..+23.
    const bf16_t* Ab = (kc < j.K1) ? (A1p + kc) : (A2p + (kc - j.K1));
    v8bf r1 = *reinterpret_cast<const v8bf*>(Ab + kb8);
    v8bf r2 = *reinterpret_cast<const v8bf*>(Ab + 16 + kb8);

    // stage next chunk into the other buffer, then wait for current chunk
    // only (ASYNCcnt completes in order; 2 ops per stage per wave).
    if (i + 1 < nch) {
      rssm_stage_panel(j.Wt, K, bcol0, kc + 32, (i & 1) ? lds0 : lds1, tid);
      asm volatile("s_wait_asynccnt 0x2" ::: "memory");
    } else {
      asm volatile("s_wait_asynccnt 0x0" ::: "memory");
    }
    __syncthreads();   // all waves' portions of current panel visible

    v16bf af;
#pragma unroll
    for (int e = 0; e < 8; ++e) { af[e] = r1[e]; af[8 + e] = r2[e]; }

    const unsigned char* pb = &panel[i & 1][0];
#pragma unroll
    for (int n = 0; n < 4; ++n) {
      // B fragment from LDS: column (n*16+l15), K kc+hi*8..+7 and kc+16+hi*8..+7
      const uint32_t boff = (uint32_t)((n * 16 + l15) * PANEL_ROW + hi * 16);
      v8bf b1 = *reinterpret_cast<const v8bf*>(pb + boff);
      v8bf b2 = *reinterpret_cast<const v8bf*>(pb + boff + 32);
      v16bf bfv;
#pragma unroll
      for (int e = 0; e < 8; ++e) { bfv[e] = b1[e]; bfv[8 + e] = b2[e]; }
      c[n] = __builtin_amdgcn_wmma_f32_16x16x32_bf16(false, af, false, bfv,
                                                     (short)0, c[n], false, false);
    }
    __syncthreads();   // all waves done reading before buffer is restaged
  }

  // C layout: VGPR d -> row (wave*16 + hi*8 + d), col bcol[n]
  const int m0 = wave * 16 + hi * 8;
#pragma unroll
  for (int n = 0; n < 4; ++n) {
#pragma unroll
    for (int d = 0; d < 8; ++d) {
      float v = c[n][d];
      if (j.elu) v = (v > 0.0f) ? v : (expf(v) - 1.0f);
      const size_t idx = (size_t)(m0 + d) * j.N + bcol[n];
      if (j.Cf)  j.Cf[idx]  = v;
      if (j.Cbf) j.Cbf[idx] = (bf16_t)v;
    }
  }
}

// ---------------------------------------------------------------------------
// GRU gate fusion: deter <- GRUCell(sa, deter). Writes f32 deter (carry),
// bf16 deter (next GEMMs' A), and mirrors into the two det output slots.
// ---------------------------------------------------------------------------
__device__ __forceinline__ float rssm_sigmoid(float x) { return 1.0f / (1.0f + expf(-x)); }

__global__ void rssm_gru(const float* __restrict__ gi, const float* __restrict__ gh,
                         float* __restrict__ deter, bf16_t* __restrict__ deter_bf,
                         float* __restrict__ outA, float* __restrict__ outB) {
  int i = blockIdx.x * blockDim.x + threadIdx.x;   // B*DETER
  int b = i >> 10;
  int jj = i & 1023;
  const float* gib = gi + (size_t)b * 3 * DETER;
  const float* ghb = gh + (size_t)b * 3 * DETER;
  float r = rssm_sigmoid(gib[jj] + ghb[jj]);
  float z = rssm_sigmoid(gib[DETER + jj] + ghb[DETER + jj]);
  float n = tanhf(gib[2 * DETER + jj] + r * ghb[2 * DETER + jj]);
  float h = deter[i];
  float d = (1.0f - z) * n + z * h;
  deter[i]    = d;
  deter_bf[i] = (bf16_t)d;
  outA[i] = d;
  outB[i] = d;
}

// ---------------------------------------------------------------------------
// Fused prior/posterior straight-through gumbel one-hot sample.
// One (b,cat) row of 32 classes per wave; shuffle-argmax across 32 lanes.
// blockIdx.y: 0 = prior, 1 = posterior (posterior also feeds next-step stoch).
// ---------------------------------------------------------------------------
__device__ __forceinline__ uint32_t rssm_hash(uint32_t x) {
  x ^= x >> 16; x *= 0x7feb352dU;
  x ^= x >> 15; x *= 0x846ca68bU;
  x ^= x >> 16;
  return x;
}

__global__ void rssm_sample2(const float* __restrict__ lg0, float* __restrict__ st0,
                             const float* __restrict__ lg1, float* __restrict__ st1,
                             bf16_t* __restrict__ stoch_next, int t) {
  const int which = blockIdx.y;
  const float* logits = which ? lg1 : lg0;
  float*       st_out = which ? st1 : st0;

  int lane = threadIdx.x & 31;
  int row  = blockIdx.x * (blockDim.x >> 5) + (threadIdx.x >> 5); // 0..B*32-1
  int idx  = row * 32 + lane;                                     // b*1024+cat*32+cls
  float lg = logits[idx];

  uint32_t h = rssm_hash(rssm_hash((uint32_t)idx * 2654435761u + 0x9e3779b9u) ^
                         rssm_hash((uint32_t)(t * 2 + which) + 0x85ebca6bu));
  float u = (float)(h & 0x00FFFFFFu) * (1.0f / 16777216.0f);
  u = fmaxf(u, 1.0e-7f);
  float g = -logf(-logf(u));

  float v = lg + g;
  int ai = lane;
#pragma unroll
  for (int off = 16; off > 0; off >>= 1) {
    float ov = __shfl_xor(v, off, 32);
    int   oi = __shfl_xor(ai, off, 32);
    if (ov > v || (ov == v && oi < ai)) { v = ov; ai = oi; }
  }
  float oh = (lane == ai) ? 1.0f : 0.0f;
  st_out[idx] = oh;
  if (which) stoch_next[idx] = (bf16_t)oh;
}

// ---------------------------------------------------------------------------
// Host-side orchestration (graph-capture safe: launches only)
// ---------------------------------------------------------------------------
extern "C" void kernel_launch(void* const* d_in, const int* in_sizes, int n_in,
                              void* d_out, int out_size, void* d_ws, size_t ws_size,
                              hipStream_t stream) {
  (void)in_sizes; (void)n_in; (void)out_size; (void)ws_size;

  const float* obs  = (const float*)d_in[1];    // [T,B,EMBED]
  const float* act  = (const float*)d_in[2];    // [T,B,ACT]
  const float* Wemb = (const float*)d_in[3];    // [1088,1024]
  const float* bemb = (const float*)d_in[4];
  const float* Wih  = (const float*)d_in[5];    // [1024,3072]
  const float* bih  = (const float*)d_in[6];
  const float* Whh  = (const float*)d_in[7];    // [1024,3072]
  const float* bhh  = (const float*)d_in[8];
  const float* Wp1  = (const float*)d_in[9];    // [1024,1024]
  const float* bp1  = (const float*)d_in[10];
  const float* Wp2  = (const float*)d_in[11];   // [1024,1024]
  const float* bp2  = (const float*)d_in[12];
  const float* Wq1  = (const float*)d_in[13];   // [2048,1024]
  const float* bq1  = (const float*)d_in[14];
  const float* Wq2  = (const float*)d_in[15];   // [1024,1024]
  const float* bq2  = (const float*)d_in[16];
  float* out = (float*)d_out;                   // [6,T,B,1024]

  // ---- workspace carve-out -------------------------------------------------
  char* ws = (char*)d_ws;
  size_t off = 0;
  auto carve = [&](size_t bytes) -> char* {
    off = (off + 255) & ~(size_t)255;
    char* p = ws + off;
    off += bytes;
    return p;
  };
  bf16_t* wt_emb = (bf16_t*)carve((size_t)(STOCH + ACT_DIM) * DETER * 2);
  bf16_t* wt_ih  = (bf16_t*)carve((size_t)DETER * 3 * DETER * 2);
  bf16_t* wt_hh  = (bf16_t*)carve((size_t)DETER * 3 * DETER * 2);
  bf16_t* wt_p1  = (bf16_t*)carve((size_t)DETER * NODE * 2);
  bf16_t* wt_p2  = (bf16_t*)carve((size_t)NODE * STOCH * 2);
  bf16_t* wt_q1  = (bf16_t*)carve((size_t)(DETER + EMBED) * NODE * 2);
  bf16_t* wt_q2  = (bf16_t*)carve((size_t)NODE * STOCH * 2);
  bf16_t* obs_bf = (bf16_t*)carve((size_t)T_STEPS * B_SIZE * EMBED * 2);
  bf16_t* act_bf = (bf16_t*)carve((size_t)T_STEPS * B_SIZE * ACT_DIM * 2);
  bf16_t* stoch_bf = (bf16_t*)carve((size_t)B_SIZE * STOCH * 2);
  bf16_t* deter_bf = (bf16_t*)carve((size_t)B_SIZE * DETER * 2);
  bf16_t* sa_bf    = (bf16_t*)carve((size_t)B_SIZE * DETER * 2);
  bf16_t* hidp_bf  = (bf16_t*)carve((size_t)B_SIZE * NODE * 2);
  bf16_t* hidq_bf  = (bf16_t*)carve((size_t)B_SIZE * NODE * 2);
  float* deter = (float*)carve((size_t)B_SIZE * DETER * 4);
  float* gi    = (float*)carve((size_t)B_SIZE * 3 * DETER * 4);
  float* gh    = (float*)carve((size_t)B_SIZE * 3 * DETER * 4);

  auto slot = [&](int s, int t) -> float* {
    return out + ((size_t)s * T_STEPS + t) * B_SIZE * 1024;
  };
  auto cdiv = [](int a, int b) { return (a + b - 1) / b; };

  // ---- prologue: weight transpose->bf16, input convert, carry init ---------
  rssm_convert_t<<<cdiv((STOCH + ACT_DIM) * DETER, 256), 256, 0, stream>>>(
      Wemb, wt_emb, STOCH + ACT_DIM, DETER);
  rssm_convert_t<<<cdiv(DETER * 3 * DETER, 256), 256, 0, stream>>>(
      Wih, wt_ih, DETER, 3 * DETER);
  rssm_convert_t<<<cdiv(DETER * 3 * DETER, 256), 256, 0, stream>>>(
      Whh, wt_hh, DETER, 3 * DETER);
  rssm_convert_t<<<cdiv(DETER * NODE, 256), 256, 0, stream>>>(Wp1, wt_p1, DETER, NODE);
  rssm_convert_t<<<cdiv(NODE * STOCH, 256), 256, 0, stream>>>(Wp2, wt_p2, NODE, STOCH);
  rssm_convert_t<<<cdiv((DETER + EMBED) * NODE, 256), 256, 0, stream>>>(
      Wq1, wt_q1, DETER + EMBED, NODE);
  rssm_convert_t<<<cdiv(NODE * STOCH, 256), 256, 0, stream>>>(Wq2, wt_q2, NODE, STOCH);
  rssm_convert<<<cdiv(T_STEPS * B_SIZE * EMBED, 256), 256, 0, stream>>>(
      obs, obs_bf, T_STEPS * B_SIZE * EMBED);
  rssm_convert<<<cdiv(T_STEPS * B_SIZE * ACT_DIM, 256), 256, 0, stream>>>(
      act, act_bf, T_STEPS * B_SIZE * ACT_DIM);
  rssm_zero_bf16<<<cdiv(B_SIZE * STOCH, 256), 256, 0, stream>>>(stoch_bf, B_SIZE * STOCH);
  rssm_zero_bf16<<<cdiv(B_SIZE * DETER, 256), 256, 0, stream>>>(deter_bf, B_SIZE * DETER);
  rssm_zero_f32<<<cdiv(B_SIZE * DETER, 256), 256, 0, stream>>>(deter, B_SIZE * DETER);

  // ---- sequential rollout: 6 launches per step -----------------------------
  for (int t = 0; t < T_STEPS; ++t) {
    const bf16_t* act_t = act_bf + (size_t)t * B_SIZE * ACT_DIM;
    const bf16_t* obs_t = obs_bf + (size_t)t * B_SIZE * EMBED;

    // 1) sa = elu([stoch | act] @ W_embed + b)          (bf16 out only)
    GemmJob je{stoch_bf, act_t, wt_emb, bemb, nullptr, sa_bf,
               STOCH, ACT_DIM, DETER, 1};
    rssm_gemm2<<<dim3(DETER / 64, 1), 128, 0, stream>>>(je, je);

    // 2) gi = sa @ W_ih + b_ih ; gh = deter @ W_hh + b_hh   (fused pair)
    GemmJob jgi{sa_bf, nullptr, wt_ih, bih, gi, nullptr, DETER, 0, 3 * DETER, 0};
    GemmJob jgh{deter_bf, nullptr, wt_hh, bhh, gh, nullptr, DETER, 0, 3 * DETER, 0};
    rssm_gemm2<<<dim3(3 * DETER / 64, 2), 128, 0, stream>>>(jgi, jgh);

    // 3) GRU gate fusion -> deter (f32 + bf16) + both det output slots
    rssm_gru<<<cdiv(B_SIZE * DETER, 256), 256, 0, stream>>>(
        gi, gh, deter, deter_bf, slot(2, t), slot(5, t));

    // 4) hidp = elu(deter @ Wp1 + bp1) ; hidq = elu([deter|obs] @ Wq1 + bq1)
    GemmJob jp1{deter_bf, nullptr, wt_p1, bp1, nullptr, hidp_bf, DETER, 0, NODE, 1};
    GemmJob jq1{deter_bf, obs_t, wt_q1, bq1, nullptr, hidq_bf, DETER, EMBED, NODE, 1};
    rssm_gemm2<<<dim3(NODE / 64, 2), 128, 0, stream>>>(jp1, jq1);

    // 5) prior/posterior logits straight into out[0,t] / out[3,t]
    GemmJob jp2{hidp_bf, nullptr, wt_p2, bp2, slot(0, t), nullptr, NODE, 0, STOCH, 0};
    GemmJob jq2{hidq_bf, nullptr, wt_q2, bq2, slot(3, t), nullptr, NODE, 0, STOCH, 0};
    rssm_gemm2<<<dim3(STOCH / 64, 2), 128, 0, stream>>>(jp2, jq2);

    // 6) fused gumbel one-hot samples; posterior sample feeds next-step stoch
    rssm_sample2<<<dim3((B_SIZE * 32) / 8, 2), 256, 0, stream>>>(
        slot(0, t), slot(1, t), slot(3, t), slot(4, t), stoch_bf, t);
  }
}
